// GroupedQueryAttention_47416438948000
// MI455X (gfx1250) — compile-verified, matrix operand order checked by reading, not run on hardware
//
#include <hip/hip_runtime.h>

// ---------------------------------------------------------------------------
// GQA attention block for MI455X (gfx1250, wave32, WMMA bf16 / f32-accum,
// async global->LDS staging tracked by ASYNCcnt)
// ---------------------------------------------------------------------------

typedef __attribute__((ext_vector_type(16))) __bf16 v16bf;
typedef __attribute__((ext_vector_type(8)))  float  v8f;
typedef __attribute__((ext_vector_type(4)))  int    v4i;

union Frag16 {           // 16 bf16 = 32 bytes = two uint4 chunks
  v16bf v;
  uint4 u[2];
};

static __device__ __forceinline__ __bf16 f2bf(float f) { return (__bf16)f; }

#define GLOBAL_AS __attribute__((address_space(1)))
#define LDS_AS    __attribute__((address_space(3)))

#if defined(__gfx1250__) && __has_builtin(__builtin_amdgcn_global_load_async_to_lds_b128)
#define HAVE_ASYNC_COPY 1
#else
#define HAVE_ASYNC_COPY 0
#endif

#if __has_builtin(__builtin_amdgcn_s_wait_asynccnt)
#define WAIT_ASYNC(n) __builtin_amdgcn_s_wait_asynccnt(n)
#else
#define WAIT_ASYNC(n) asm volatile("s_wait_asynccnt " #n ::: "memory")
#endif

// Copy one 16-byte chunk global -> LDS (async when available)
static __device__ __forceinline__ void cp_b128(const __bf16* g, __bf16* l) {
#if HAVE_ASYNC_COPY
  __builtin_amdgcn_global_load_async_to_lds_b128(
      (GLOBAL_AS v4i*)g, (LDS_AS v4i*)l, 0, 0);
#else
  *(uint4*)l = *(const uint4*)g;
#endif
}

// ---------------------------------------------------------------------------
// Tiled GEMM:  C[M,N] = A[M,K](bf16) * B[N,K](bf16)^T
// Block: 256 threads = 8 waves; tile 128x128x32; wave tile 64x32 (4x2 wmma)
// Double-buffered LDS stages filled by async copies.
// ---------------------------------------------------------------------------
#define GBM 128
#define GBN 128
#define GBK 32

// Stage a 128x32 bf16 tile: 2 x b128 per thread (4 KB per 256-thread round)
static __device__ __forceinline__ void stage_tile(const __bf16* g, int ld,
                                                  __bf16* l, int tid, int k0) {
#pragma unroll
  for (int i = 0; i < 2; ++i) {
    int c   = tid * 2 + i;                 // 0..511
    int row = c >> 2;
    int off = (c & 3) * 8;
    cp_b128(g + (size_t)row * ld + k0 + off, l + row * GBK + off);
  }
}

__global__ __launch_bounds__(256)
void gemm_bf16(const __bf16* __restrict__ A, const __bf16* __restrict__ B,
               float* __restrict__ C, int M, int N, int K) {
  __shared__ __align__(16) __bf16 sA[2][GBM][GBK];
  __shared__ __align__(16) __bf16 sB[2][GBN][GBK];

  const int tid      = threadIdx.x;
  const int lane     = tid & 31;
  const int wave     = tid >> 5;
  const int laneRow  = lane & 15;
  const int laneHalf = lane >> 4;
  const int wm = wave & 1;           // 0..1 -> 64 rows each
  const int wn = wave >> 1;          // 0..3 -> 32 cols each
  const int blockM = blockIdx.y * GBM;
  const int blockN = blockIdx.x * GBN;
  const __bf16* Ab = A + (size_t)blockM * K;
  const __bf16* Bb = B + (size_t)blockN * K;

  v8f acc[4][2];
#pragma unroll
  for (int mt = 0; mt < 4; ++mt)
#pragma unroll
    for (int nt = 0; nt < 2; ++nt)
#pragma unroll
      for (int i = 0; i < 8; ++i) acc[mt][nt][i] = 0.0f;

  const int nk = K / GBK;
  // Prologue: fill stage 0 (4 async ops per thread)
  stage_tile(Ab, K, &sA[0][0][0], tid, 0);
  stage_tile(Bb, K, &sB[0][0][0], tid, 0);

  int st = 0;
  for (int ki = 0; ki < nk; ++ki) {
    if (ki + 1 < nk) {
      // Issue next stage while current computes (4 more async ops)
      stage_tile(Ab, K, &sA[st ^ 1][0][0], tid, (ki + 1) * GBK);
      stage_tile(Bb, K, &sB[st ^ 1][0][0], tid, (ki + 1) * GBK);
      WAIT_ASYNC(4);     // oldest 4 (current stage) complete, 4 in flight
    } else {
      WAIT_ASYNC(0);
    }
    __syncthreads();

    Frag16 af[4], bf[2];
#pragma unroll
    for (int mt = 0; mt < 4; ++mt) {
      int r = wm * 64 + mt * 16 + laneRow;
      af[mt].u[0] = *(const uint4*)&sA[st][r][laneHalf * 8];
      af[mt].u[1] = *(const uint4*)&sA[st][r][16 + laneHalf * 8];
    }
#pragma unroll
    for (int nt = 0; nt < 2; ++nt) {
      int r = wn * 32 + nt * 16 + laneRow;
      bf[nt].u[0] = *(const uint4*)&sB[st][r][laneHalf * 8];
      bf[nt].u[1] = *(const uint4*)&sB[st][r][16 + laneHalf * 8];
    }
#pragma unroll
    for (int mt = 0; mt < 4; ++mt)
#pragma unroll
      for (int nt = 0; nt < 2; ++nt)
        acc[mt][nt] = __builtin_amdgcn_wmma_f32_16x16x32_bf16(
            false, af[mt].v, false, bf[nt].v, (short)0, acc[mt][nt],
            false, false);
    __syncthreads();     // all waves done with stage st before it is refilled
    st ^= 1;
  }

#pragma unroll
  for (int mt = 0; mt < 4; ++mt)
#pragma unroll
    for (int nt = 0; nt < 2; ++nt)
#pragma unroll
      for (int r = 0; r < 8; ++r) {
        int row = blockM + wm * 64 + mt * 16 + laneHalf * 8 + r;
        int col = blockN + wn * 32 + nt * 16 + laneRow;
        C[(size_t)row * N + col] = acc[mt][nt][r];
      }
}

// ---------------------------------------------------------------------------
// Elementwise converts
// ---------------------------------------------------------------------------
__global__ void f32_to_bf16_kernel(const float* __restrict__ src,
                                   __bf16* __restrict__ dst, int n) {
  int i = blockIdx.x * blockDim.x + threadIdx.x;
  if (i < n) dst[i] = f2bf(src[i]);
}

// v_f32 [T][N_KV*128] -> v_bf16 [N_KV][T][128]
__global__ void v_remap_kernel(const float* __restrict__ src,
                               __bf16* __restrict__ dst, int T) {
  int i = blockIdx.x * blockDim.x + threadIdx.x;
  int d = i & 127;
  int g = (i >> 7) & 7;
  int t = i >> 10;
  if (t < T)
    dst[((size_t)g * T + t) * 128 + d] = f2bf(src[i]);
}

// ---------------------------------------------------------------------------
// RoPE: src f32 [T][H*128] -> dst bf16 [H][T][128], 64 threads / (t,h)
// ---------------------------------------------------------------------------
__global__ void rope_kernel(const float* __restrict__ src,
                            const float* __restrict__ cos_t,
                            const float* __restrict__ sin_t,
                            const int* __restrict__ start_pos,
                            __bf16* __restrict__ dst, int H, int T) {
  int blk = blockIdx.x;
  int h = blk % H;
  int t = blk / H;
  int d = threadIdx.x;            // 0..63
  int sp = start_pos[0];
  float c = cos_t[(size_t)(sp + t) * 64 + d];
  float s = sin_t[(size_t)(sp + t) * 64 + d];
  const float* row = src + (size_t)t * (H * 128) + (size_t)h * 128;
  float x1 = row[d];
  float x2 = row[d + 64];
  __bf16* out = dst + ((size_t)h * T + t) * 128;
  out[d]      = f2bf(x1 * c - x2 * s);
  out[d + 64] = f2bf(x2 * c + x1 * s);
}

// ---------------------------------------------------------------------------
// Flash attention (causal, GQA): one block = one head x 64 query rows.
// 4 waves, each wave owns 16 query rows. Q/K/V bf16, f32 online softmax.
// K tile staged by async copy; V transposed through VGPRs.
// ---------------------------------------------------------------------------
__global__ __launch_bounds__(128)
void attn_kernel(const __bf16* __restrict__ Q, const __bf16* __restrict__ Kg,
                 const __bf16* __restrict__ Vg, __bf16* __restrict__ Y,
                 int T) {
  __shared__ __align__(16) __bf16 sK[64][128];    // keys row-major
  __shared__ __align__(16) __bf16 sVT[128][64];   // V transposed (d x key)
  __shared__ __align__(16) __bf16 sP[4][16][64];  // per-wave P staging

  const int HD = 128;
  const int h = blockIdx.y;
  const int g = h >> 2;                 // kv group (N_REP = 4)
  const int qBase = blockIdx.x * 64;
  const int tid = threadIdx.x;
  const int lane = tid & 31;
  const int wave = tid >> 5;
  const int laneRow = lane & 15;
  const int laneHalf = lane >> 4;
  const int qRow0 = qBase + wave * 16;
  const float scale = 0.08838834764831845f;   // 1/sqrt(128)

  // Q fragments (A operand), resident for the whole block
  Frag16 qf[4];
  const __bf16* qh = Q + ((size_t)h * T + qRow0) * HD;
#pragma unroll
  for (int ks = 0; ks < 4; ++ks) {
    const __bf16* p = qh + (size_t)laneRow * HD + ks * 32 + laneHalf * 8;
    qf[ks].u[0] = *(const uint4*)p;
    qf[ks].u[1] = *(const uint4*)(p + 16);
  }

  v8f o[8];
  float mrow[8], lrow[8];
#pragma unroll
  for (int dt = 0; dt < 8; ++dt)
#pragma unroll
    for (int i = 0; i < 8; ++i) o[dt][i] = 0.0f;
#pragma unroll
  for (int r = 0; r < 8; ++r) { mrow[r] = -3.0e38f; lrow[r] = 0.0f; }

  const int nkb = blockIdx.x + 1;   // causal: only key blocks <= query block
  for (int kb = 0; kb < nkb; ++kb) {
    const int kBase = kb * 64;
    const __bf16* kp = Kg + ((size_t)g * T + kBase) * HD;
    const __bf16* vp = Vg + ((size_t)g * T + kBase) * HD;

    // Stage K (async, row-major) and V (transposed via VGPRs) into LDS
#pragma unroll
    for (int i = 0; i < 8; ++i) {
      int c = tid + i * 128;          // 0..1023
      int row = c >> 4;
      int off = (c & 15) * 8;
      cp_b128(&kp[(size_t)row * HD + off], &sK[row][off]);
      uint4 u = *(const uint4*)&vp[(size_t)row * HD + off];
      const __bf16* e = (const __bf16*)&u;
#pragma unroll
      for (int j = 0; j < 8; ++j) sVT[off + j][row] = e[j];
    }
    WAIT_ASYNC(0);
    __syncthreads();

    // S = Q * K^T  (16x64 per wave, 16 WMMAs)
    v8f s[4];
#pragma unroll
    for (int nt = 0; nt < 4; ++nt) {
      v8f acc;
#pragma unroll
      for (int i = 0; i < 8; ++i) acc[i] = 0.0f;
#pragma unroll
      for (int ks = 0; ks < 4; ++ks) {
        Frag16 b;
        const __bf16* p = &sK[nt * 16 + laneRow][ks * 32 + laneHalf * 8];
        b.u[0] = *(const uint4*)p;
        b.u[1] = *(const uint4*)(p + 16);
        acc = __builtin_amdgcn_wmma_f32_16x16x32_bf16(
            false, qf[ks].v, false, b.v, (short)0, acc, false, false);
      }
      s[nt] = acc;
    }

    // Scale + causal mask + row-max (row spans 16 lanes of a half)
    float rmax[8];
#pragma unroll
    for (int r = 0; r < 8; ++r) rmax[r] = -3.0e38f;
#pragma unroll
    for (int nt = 0; nt < 4; ++nt) {
      int col = kBase + nt * 16 + laneRow;
#pragma unroll
      for (int r = 0; r < 8; ++r) {
        int qrow = qRow0 + laneHalf * 8 + r;
        float v = s[nt][r] * scale;
        v = (col <= qrow) ? v : -3.0e38f;
        s[nt][r] = v;
        rmax[r] = fmaxf(rmax[r], v);
      }
    }
#pragma unroll
    for (int r = 0; r < 8; ++r)
#pragma unroll
      for (int m = 1; m < 16; m <<= 1)
        rmax[r] = fmaxf(rmax[r], __shfl_xor(rmax[r], m, 32));

    // Online softmax update
    float corr[8], psum[8];
#pragma unroll
    for (int r = 0; r < 8; ++r) {
      float mnew = fmaxf(mrow[r], rmax[r]);
      corr[r] = __expf(mrow[r] - mnew);
      mrow[r] = mnew;
      lrow[r] *= corr[r];
      psum[r] = 0.0f;
    }
#pragma unroll
    for (int nt = 0; nt < 4; ++nt)
#pragma unroll
      for (int r = 0; r < 8; ++r) {
        float p = __expf(s[nt][r] - mrow[r]);
        s[nt][r] = p;
        psum[r] += p;
      }
#pragma unroll
    for (int r = 0; r < 8; ++r) {
#pragma unroll
      for (int m = 1; m < 16; m <<= 1) psum[r] += __shfl_xor(psum[r], m, 32);
      lrow[r] += psum[r];
    }
#pragma unroll
    for (int dt = 0; dt < 8; ++dt)
#pragma unroll
      for (int r = 0; r < 8; ++r) o[dt][r] *= corr[r];

    // C-layout -> A-layout via per-wave LDS staging (bf16)
#pragma unroll
    for (int nt = 0; nt < 4; ++nt)
#pragma unroll
      for (int r = 0; r < 8; ++r)
        sP[wave][laneHalf * 8 + r][nt * 16 + laneRow] = f2bf(s[nt][r]);
    asm volatile("s_wait_dscnt 0" ::: "memory");  // same-wave LDS RAW fence

    Frag16 pa[2];
#pragma unroll
    for (int ks = 0; ks < 2; ++ks) {
      const __bf16* p = &sP[wave][laneRow][ks * 32 + laneHalf * 8];
      pa[ks].u[0] = *(const uint4*)p;
      pa[ks].u[1] = *(const uint4*)(p + 16);
    }
    // O += P * V   (16x128 per wave, 16 WMMAs)
#pragma unroll
    for (int dt = 0; dt < 8; ++dt)
#pragma unroll
      for (int ks = 0; ks < 2; ++ks) {
        Frag16 b;
        const __bf16* p = &sVT[dt * 16 + laneRow][ks * 32 + laneHalf * 8];
        b.u[0] = *(const uint4*)p;
        b.u[1] = *(const uint4*)(p + 16);
        o[dt] = __builtin_amdgcn_wmma_f32_16x16x32_bf16(
            false, pa[ks].v, false, b.v, (short)0, o[dt], false, false);
      }
    __syncthreads();
  }

  // Finalize: divide by row sum, write bf16 in [T][H*128] layout
#pragma unroll
  for (int dt = 0; dt < 8; ++dt)
#pragma unroll
    for (int r = 0; r < 8; ++r) {
      int qrow = qRow0 + laneHalf * 8 + r;
      int d = dt * 16 + laneRow;
      Y[(size_t)qrow * 4096 + (size_t)h * 128 + d] =
          f2bf(o[dt][r] / lrow[r]);
    }
}

// ---------------------------------------------------------------------------
// Launcher
// ---------------------------------------------------------------------------
extern "C" void kernel_launch(void* const* d_in, const int* in_sizes, int n_in,
                              void* d_out, int out_size, void* d_ws,
                              size_t ws_size, hipStream_t stream) {
  const int T = 2048, DIM = 4096, NH = 32, NKV = 8;

  const float* x   = (const float*)d_in[0];
  const float* cs  = (const float*)d_in[1];
  const float* sn  = (const float*)d_in[2];
  const float* wq  = (const float*)d_in[3];
  const float* wk  = (const float*)d_in[4];
  const float* wv  = (const float*)d_in[5];
  const float* wo  = (const float*)d_in[6];
  const int* spos  = (const int*)d_in[7];
  float* out = (float*)d_out;

  char* ws = (char*)d_ws;
  size_t off = 0;
  __bf16* x_bf  = (__bf16*)(ws + off); off += (size_t)T * DIM * 2;        // 16 MB
  __bf16* wq_bf = (__bf16*)(ws + off); off += (size_t)DIM * DIM * 2;      // 32 MB
  __bf16* wk_bf = (__bf16*)(ws + off); off += (size_t)NKV * 128 * DIM * 2;//  8 MB
  __bf16* wv_bf = (__bf16*)(ws + off); off += (size_t)NKV * 128 * DIM * 2;//  8 MB
  __bf16* wo_bf = (__bf16*)(ws + off); off += (size_t)DIM * DIM * 2;      // 32 MB
  float*  q_f32 = (float*)(ws + off);  off += (size_t)T * DIM * 4;        // 32 MB
  float*  k_f32 = (float*)(ws + off);  off += (size_t)T * NKV * 128 * 4;  //  8 MB
  float*  v_f32 = (float*)(ws + off);  off += (size_t)T * NKV * 128 * 4;  //  8 MB
  __bf16* q_bf  = (__bf16*)(ws + off); off += (size_t)T * DIM * 2;        // 16 MB
  __bf16* k_bf  = (__bf16*)(ws + off); off += (size_t)T * NKV * 128 * 2;  //  4 MB
  __bf16* v_bf  = (__bf16*)(ws + off); off += (size_t)T * NKV * 128 * 2;  //  4 MB
  __bf16* y_bf  = (__bf16*)q_f32;  // alias: q_f32 is dead after RoPE

  // 1) Convert activations + weights to bf16 (one-time, bandwidth-trivial)
  {
    int n = T * DIM;
    f32_to_bf16_kernel<<<(n + 255) / 256, 256, 0, stream>>>(x, x_bf, n);
    n = DIM * DIM;
    f32_to_bf16_kernel<<<(n + 255) / 256, 256, 0, stream>>>(wq, wq_bf, n);
    f32_to_bf16_kernel<<<(n + 255) / 256, 256, 0, stream>>>(wo, wo_bf, n);
    n = NKV * 128 * DIM;
    f32_to_bf16_kernel<<<(n + 255) / 256, 256, 0, stream>>>(wk, wk_bf, n);
    f32_to_bf16_kernel<<<(n + 255) / 256, 256, 0, stream>>>(wv, wv_bf, n);
  }
  // 2) QKV projections (WMMA GEMM, async double-buffered)
  gemm_bf16<<<dim3(DIM / GBN, T / GBM), 256, 0, stream>>>(
      x_bf, wq_bf, q_f32, T, DIM, DIM);
  gemm_bf16<<<dim3(NKV * 128 / GBN, T / GBM), 256, 0, stream>>>(
      x_bf, wk_bf, k_f32, T, NKV * 128, DIM);
  gemm_bf16<<<dim3(NKV * 128 / GBN, T / GBM), 256, 0, stream>>>(
      x_bf, wv_bf, v_f32, T, NKV * 128, DIM);
  // 3) RoPE (writes [head][T][128] bf16)
  rope_kernel<<<T * NH, 64, 0, stream>>>(q_f32, cs, sn, spos, q_bf, NH, T);
  rope_kernel<<<T * NKV, 64, 0, stream>>>(k_f32, cs, sn, spos, k_bf, NKV, T);
  // 4) V -> bf16 [g][T][128]
  {
    int n = T * NKV * 128;
    v_remap_kernel<<<(n + 255) / 256, 256, 0, stream>>>(v_f32, v_bf, T);
  }
  // 5) Flash attention (causal, GQA)  -- y_bf aliases q_f32 (now dead)
  attn_kernel<<<dim3(T / 64, NH), 128, 0, stream>>>(q_bf, k_bf, v_bf, y_bf, T);
  // 6) Output projection
  gemm_bf16<<<dim3(DIM / GBN, T / GBM), 256, 0, stream>>>(
      y_bf, wo_bf, out, T, DIM, DIM);
}
